// MambaG2G_1443109011831
// MI455X (gfx1250) — compile-verified, matrix operand order checked by reading, not run on hardware
//
#include <hip/hip_runtime.h>

// ---------------------------------------------------------------------------
// Mamba-style block for MI455X (gfx1250, wave32, WMMA bf16 16x16x32 + TDM).
// Only e[:, -1, :] is needed for outputs (x, mu); scan state computed for all t.
// ---------------------------------------------------------------------------

#define B_   32
#define L_   256
#define DM_  512
#define ED_  1024
#define N_   16
#define DC_  4
#define DTR_ 32
#define DIN_ 512
#define DOUT_ 64
#define ROWS_ (B_ * L_)   // 8192

typedef __attribute__((ext_vector_type(16))) __bf16    v16bf;
typedef __attribute__((ext_vector_type(8)))  float     v8f;
typedef __attribute__((ext_vector_type(4)))  unsigned  u32x4;
typedef __attribute__((ext_vector_type(8)))  int       i32x8;
typedef __attribute__((ext_vector_type(4)))  int       i32x4;

// ---------------------------------------------------------------------------
// Global-memory fragment loader: lanes 0-15 -> rows r0..r0+15, k0..k0+15;
// lanes 16-31 -> same rows, k0+16..k0+31 (one 32B contiguous load per lane).
// ---------------------------------------------------------------------------
__device__ __forceinline__ v16bf load_frag16(const __bf16* __restrict__ base,
                                             int ld, int r0, int k0) {
  const int lane = threadIdx.x & 31;
  const __bf16* p = base + (size_t)(r0 + (lane & 15)) * ld + (k0 + ((lane >> 4) << 4));
  return *(const v16bf*)p;
}

// LDS A-strip layout: element i = r*K + c stored at byte 2*i + 16*(i>>9)
// (TDM pad: 4 DWORDs inserted every 256 DWORDs -> bank-spread rows, 16B align).
__device__ __forceinline__ v16bf lds_fragA(const unsigned char* sA, int K, int kt) {
  const int lane = threadIdx.x & 31;
  const int i = (lane & 15) * K + kt * 32 + ((lane >> 4) << 4);
  return *(const v16bf*)(sA + 2 * i + ((i >> 9) << 4));
}

__device__ __forceinline__ v8f wmma_bf16(v16bf a, v16bf b, v8f c) {
  return __builtin_amdgcn_wmma_f32_16x16x32_bf16(false, a, false, b, (short)0, c,
                                                 false, false);
}

__device__ __forceinline__ float silu_f(float v) {
  return v / (1.f + __expf(-v));
}

// ---------------------------------------------------------------------------
// TDM: stage a 16-row x K-col bf16 tile (row stride lda) into LDS offset 0.
// D# built per CDNA5 ISA ch.8: 2D tile, data_size=2B, pad 4DW per 256DW.
// Issued by wave 0 only (TENSORcnt-tracked); callers barrier afterwards.
// ---------------------------------------------------------------------------
__device__ __forceinline__ void tdm_stage_A(const __bf16* A, int lda, int m0, int K) {
#if defined(__has_builtin) && __has_builtin(__builtin_amdgcn_tensor_load_to_lds)
  if (threadIdx.x < 32) {
    unsigned long long ga = (unsigned long long)(const void*)(A + (size_t)m0 * lda);
    // group0: count=1 | lds_addr | global_addr[56:0] | type=2 ("image")
    u32x4 g0 = { 1u, 0u, (unsigned)ga,
                 ((unsigned)(ga >> 32) & 0x01FFFFFFu) | (2u << 30) };
    // group1: data_size=2B, pad_enable, pad_interval=256DW, pad_amount=4DW,
    //         tensor_dim0=K, tensor_dim1=16, tile_dim0=K, tile_dim1=16,
    //         tensor_dim0_stride=lda
    int w0 = (1 << 16) | (1 << 20) | (7 << 22) | (3 << 25);
    i32x8 g1 = { w0,
                 (int)((K & 0xFFFF) << 16),
                 (int)((K >> 16) | (16 << 16)),
                 (int)((K & 0xFFFF) << 16),
                 16, lda, 0, 0 };
    i32x4 gz = { 0, 0, 0, 0 };
#if __clang_major__ >= 23
    i32x8 ge = { 0, 0, 0, 0, 0, 0, 0, 0 };
    __builtin_amdgcn_tensor_load_to_lds(g0, g1, gz, gz, ge, 0);
#else
    __builtin_amdgcn_tensor_load_to_lds(g0, g1, gz, gz, 0);
#endif
#if defined(__has_builtin) && __has_builtin(__builtin_amdgcn_s_wait_tensorcnt)
    __builtin_amdgcn_s_wait_tensorcnt(0);
#else
    asm volatile("s_wait_tensorcnt 0x0" ::: "memory");
#endif
  }
#else
  (void)A; (void)lda; (void)m0; (void)K;   // fallback handled by caller
#endif
}

// ---------------------------------------------------------------------------
// f32 -> bf16 conversion
// ---------------------------------------------------------------------------
__global__ void f2bf_kernel(const float* __restrict__ s, __bf16* __restrict__ d, int n) {
  int i = blockIdx.x * blockDim.x + threadIdx.x;
  if (i < n) d[i] = (__bf16)s[i];
}

// ---------------------------------------------------------------------------
// Kernel A: e = input @ W_enc^T + b_enc ; u = rmsnorm(e) * rms_w  (bf16 out)
// Also captures e_last = e[:, L-1, :] (f32).
// ---------------------------------------------------------------------------
__global__ void encode_rmsnorm_kernel(const __bf16* __restrict__ inb,
                                      const __bf16* __restrict__ Wenc,
                                      const float*  __restrict__ b_enc,
                                      const float*  __restrict__ rms_w,
                                      __bf16* __restrict__ u,
                                      float*  __restrict__ e_last) {
  __shared__ float rowsum[16];
  const int m0   = blockIdx.x * 16;
  const int wave = threadIdx.x >> 5;
  const int lane = threadIdx.x & 31;
  const int half = lane >> 4;

  v8f acc[4] = {};
  // register-pipelined K loop: load kt+1 while WMMAs consume kt
  v16bf a = load_frag16(inb, DM_, m0, 0);
#pragma unroll 4
  for (int kt = 0; kt < DM_ / 32; ++kt) {
    v16bf an = a;
    if (kt + 1 < DM_ / 32) an = load_frag16(inb, DM_, m0, (kt + 1) * 32);
#pragma unroll
    for (int j = 0; j < 4; ++j) {
      v16bf bw = load_frag16(Wenc, DM_, (wave * 4 + j) * 16, kt * 32);
      acc[j] = wmma_bf16(a, bw, acc[j]);
    }
    a = an;
  }

  if (threadIdx.x < 16) rowsum[threadIdx.x] = 0.f;
  __syncthreads();

  // bias + accumulate sum of squares per row (LDS float atomics: ds_add_f32)
#pragma unroll
  for (int j = 0; j < 4; ++j) {
    int n = wave * 64 + j * 16 + (lane & 15);
    float bv = b_enc[n];
#pragma unroll
    for (int r = 0; r < 8; ++r) {
      float v = acc[j][r] + bv;
      acc[j][r] = v;
      atomicAdd(&rowsum[r + half * 8], v * v);
    }
  }
  __syncthreads();

#pragma unroll
  for (int r = 0; r < 8; ++r) {
    int row = r + half * 8;
    int m   = m0 + row;
    float rs = rsqrtf(rowsum[row] * (1.f / (float)DM_) + 1e-5f);
#pragma unroll
    for (int j = 0; j < 4; ++j) {
      int n = wave * 64 + j * 16 + (lane & 15);
      float ev = acc[j][r];
      u[(size_t)m * DM_ + n] = (__bf16)(ev * rs * rms_w[n]);
      if ((m & (L_ - 1)) == (L_ - 1))
        e_last[(size_t)(m >> 8) * DM_ + n] = ev;   // m>>8 == batch index
    }
  }
}

// ---------------------------------------------------------------------------
// Generic WMMA GEMM: out[m,n] = sum_k A[m,k] * W[n,k]   (A: lda, W: ldw == K)
// A strip (16 x K) staged once per block in LDS via the Tensor Data Mover;
// W fragments stream from global (weights are L2-resident).
// grid.x = M/16, grid.y = ceil(N/128); 8 waves, one 16x16 tile each.
// Epilogue modes:
//   0: (+bias) f32        1: softplus(acc+bias) f32
//   2: silu(acc+bias) f32 + bf16 copy
//   3: (+bias) f32 + bf16 copy
//   4: acc + aux[m*N+n] f32 + bf16 copy
// ---------------------------------------------------------------------------
__global__ void gemm_bf16_wmma(const __bf16* __restrict__ A, int lda,
                               const __bf16* __restrict__ W,
                               const float*  __restrict__ bias,
                               const float*  __restrict__ aux,
                               float*  __restrict__ outf,
                               __bf16* __restrict__ outb,
                               int N, int K, int mode) {
  // Sole __shared__ object of this kernel -> LDS offset 0 (matches D#.lds_addr).
  __shared__ __align__(32) unsigned char sA[33280];   // 16 x 1024 bf16 + pads
  const int wave = threadIdx.x >> 5;
  const int lane = threadIdx.x & 31;
  const int m0 = blockIdx.x * 16;
  const int nt = blockIdx.y * 8 + wave;

#if defined(__has_builtin) && __has_builtin(__builtin_amdgcn_tensor_load_to_lds)
  tdm_stage_A(A, lda, m0, K);
#else
  // cooperative fallback staging with the same padded layout
  for (int i = threadIdx.x * 4; i < 16 * K; i += 1024) {
    int r = i / K, c = i - r * K;
    *(unsigned long long*)(sA + 2 * i + ((i >> 9) << 4)) =
        *(const unsigned long long*)(A + (size_t)(m0 + r) * lda + c);
  }
#endif
  __syncthreads();
  if (nt * 16 >= N) return;
  const int n0 = nt * 16;

  const int nk = K / 32;
  v8f acc = {};
  // register-pipelined: fetch kt+1 fragments while WMMA consumes kt
  v16bf a = lds_fragA(sA, K, 0);
  v16bf b = load_frag16(W, K, n0, 0);
  for (int kt = 0; kt < nk; ++kt) {
    v16bf a2 = a, b2 = b;
    if (kt + 1 < nk) {
      a2 = lds_fragA(sA, K, kt + 1);
      b2 = load_frag16(W, K, n0, (kt + 1) * 32);
    }
    acc = wmma_bf16(a, b, acc);
    a = a2; b = b2;
  }

  const int n    = n0 + (lane & 15);
  const int half = lane >> 4;
  const float bv = bias ? bias[n] : 0.f;
#pragma unroll
  for (int r = 0; r < 8; ++r) {
    int m = m0 + r + half * 8;
    float v = acc[r] + bv;
    if (mode == 1) v = (v > 20.f) ? v : log1pf(__expf(v));        // softplus
    else if (mode == 2) v = silu_f(v);
    else if (mode == 4) v = acc[r] + aux[(size_t)m * N + n];
    outf[(size_t)m * N + n] = v;
    if (mode >= 2 && outb) outb[(size_t)m * N + n] = (__bf16)v;
  }
}

// ---------------------------------------------------------------------------
// Depthwise causal conv (DC=4) + bias + SiLU. One thread per (b,t,c).
// ---------------------------------------------------------------------------
__global__ void conv_silu_kernel(const float* __restrict__ xpre,
                                 const float* __restrict__ cw,
                                 const float* __restrict__ cb,
                                 float* __restrict__ xs,
                                 __bf16* __restrict__ xsbf) {
  size_t idx = (size_t)blockIdx.x * blockDim.x + threadIdx.x;
  if (idx >= (size_t)ROWS_ * ED_) return;
  int c   = (int)(idx % ED_);
  int row = (int)(idx / ED_);          // row = b*L + t
  int t   = row & (L_ - 1);
  float a = cb[c];
#pragma unroll
  for (int j = 0; j < DC_; ++j) {
    int tt = t - (DC_ - 1) + j;
    if (tt >= 0)
      a += cw[c * DC_ + j] * xpre[(size_t)(row - (DC_ - 1) + j) * ED_ + c];
  }
  float v = silu_f(a);
  xs[idx]   = v;
  xsbf[idx] = (__bf16)v;
}

// ---------------------------------------------------------------------------
// Selective scan: one thread per (b,e), N=16 states in registers.
// dA/dBu fused in-loop (never materialized); y emitted only at t = L-1.
// Next-timestep rows prefetched (global_prefetch_b8).
// ---------------------------------------------------------------------------
__global__ void scan_kernel(const float* __restrict__ delta,
                            const float* __restrict__ xs,
                            const float* __restrict__ dbc,
                            const float* __restrict__ A_log,
                            const float* __restrict__ D_,
                            float* __restrict__ ylast) {
  __shared__ float sBC[32];
  const int b = blockIdx.x >> 2;
  const int e = (blockIdx.x & 3) * 256 + threadIdx.x;

  float an[N_];
#pragma unroll
  for (int n = 0; n < N_; ++n) an[n] = -__expf(A_log[(size_t)e * N_ + n]);
  const float de = D_[e];

  float h[N_] = {};
  for (int t = 0; t < L_; ++t) {
    const size_t row = (size_t)b * L_ + t;
    if (threadIdx.x < 32) sBC[threadIdx.x] = dbc[row * 64 + DTR_ + threadIdx.x];
    __syncthreads();
    const size_t idx = row * ED_ + e;
    const float dv  = delta[idx];
    const float xv  = xs[idx];
    if (t + 1 < L_) {
      __builtin_prefetch(&delta[idx + ED_], 0, 1);
      __builtin_prefetch(&xs[idx + ED_], 0, 1);
    }
    const float dbx = dv * xv;
#pragma unroll
    for (int n = 0; n < N_; ++n)
      h[n] = __expf(dv * an[n]) * h[n] + dbx * sBC[n];
    if (t == L_ - 1) {
      float y = 0.f;
#pragma unroll
      for (int n = 0; n < N_; ++n) y += h[n] * sBC[16 + n];
      ylast[(size_t)b * ED_ + e] = y + de * xv;
    }
    __syncthreads();
  }
}

// ---------------------------------------------------------------------------
// Last-timestep gate: z = u_last @ W_in[ED:].T ; ybf = bf16(y * silu(z)).
// ---------------------------------------------------------------------------
__global__ void zgate_kernel(const __bf16* __restrict__ u,
                             const __bf16* __restrict__ Win,
                             const float*  __restrict__ ylast,
                             __bf16* __restrict__ ybf) {
  int i = blockIdx.x * blockDim.x + threadIdx.x;   // i = b*ED + e
  int b = i >> 10, e = i & (ED_ - 1);
  const __bf16* ur = u   + (size_t)(b * L_ + (L_ - 1)) * DM_;
  const __bf16* wr = Win + (size_t)(ED_ + e) * DM_;
  float z = 0.f;
  for (int k = 0; k < DM_; ++k) z += (float)ur[k] * (float)wr[k];
  ybf[i] = (__bf16)(ylast[i] * silu_f(z));
}

// ---------------------------------------------------------------------------
// Host side
// ---------------------------------------------------------------------------
extern "C" void kernel_launch(void* const* d_in, const int* in_sizes, int n_in,
                              void* d_out, int out_size, void* d_ws, size_t ws_size,
                              hipStream_t stream) {
  const float* input   = (const float*)d_in[0];
  const float* W_enc   = (const float*)d_in[1];
  const float* b_enc   = (const float*)d_in[2];
  const float* rms_w   = (const float*)d_in[3];
  const float* W_in    = (const float*)d_in[4];
  const float* conv_w  = (const float*)d_in[5];
  const float* conv_b  = (const float*)d_in[6];
  const float* W_x     = (const float*)d_in[7];
  const float* W_dt    = (const float*)d_in[8];
  const float* b_dt    = (const float*)d_in[9];
  const float* A_log   = (const float*)d_in[10];
  const float* D_      = (const float*)d_in[11];
  const float* W_out   = (const float*)d_in[12];
  const float* W_outfc = (const float*)d_in[13];
  const float* b_outfc = (const float*)d_in[14];
  const float* W_mu    = (const float*)d_in[15];
  const float* b_mu    = (const float*)d_in[16];
  float* out = (float*)d_out;

  char* ws = (char*)d_ws;
  size_t off = 0;
  auto alloc = [&](size_t bytes) { char* p = ws + off; off += (bytes + 255) & ~(size_t)255; return p; };

  __bf16* in_bf   = (__bf16*)alloc((size_t)ROWS_ * DM_ * 2);
  __bf16* Wenc_bf = (__bf16*)alloc((size_t)DM_ * DM_ * 2);
  __bf16* Win_bf  = (__bf16*)alloc((size_t)2 * ED_ * DM_ * 2);
  __bf16* Wx_bf   = (__bf16*)alloc((size_t)64 * ED_ * 2);
  __bf16* Wdt_bf  = (__bf16*)alloc((size_t)ED_ * DTR_ * 2);
  __bf16* Wout_bf = (__bf16*)alloc((size_t)DM_ * ED_ * 2);
  __bf16* Wofc_bf = (__bf16*)alloc((size_t)DIN_ * DM_ * 2);
  __bf16* Wmu_bf  = (__bf16*)alloc((size_t)DOUT_ * DIN_ * 2);
  __bf16* u_bf    = (__bf16*)alloc((size_t)ROWS_ * DM_ * 2);
  float*  e_last  = (float*) alloc((size_t)B_ * DM_ * 4);
  float*  x_pre   = (float*) alloc((size_t)ROWS_ * ED_ * 4);  // reused for delta
  float*  xs      = (float*) alloc((size_t)ROWS_ * ED_ * 4);
  __bf16* xs_bf   = (__bf16*)alloc((size_t)ROWS_ * ED_ * 2);
  float*  dbc     = (float*) alloc((size_t)ROWS_ * 64 * 4);
  __bf16* dbc_bf  = (__bf16*)alloc((size_t)ROWS_ * 64 * 2);
  float*  ylast   = (float*) alloc((size_t)B_ * ED_ * 4);
  __bf16* y_bf    = (__bf16*)alloc((size_t)B_ * ED_ * 2);
  float*  eout    = (float*) alloc((size_t)B_ * DM_ * 4);
  __bf16* eout_bf = (__bf16*)alloc((size_t)B_ * DM_ * 2);
  __bf16* x_bf    = (__bf16*)alloc((size_t)B_ * DIN_ * 2);
  float*  delta   = x_pre;   // x_pre dead after conv
  (void)in_sizes; (void)n_in; (void)out_size; (void)ws_size;

  const dim3 T(256);
  auto cvt = [&](const float* s, __bf16* d, int n) {
    f2bf_kernel<<<dim3((n + 255) / 256), T, 0, stream>>>(s, d, n);
  };

  // 1) precision conversion (activations + all GEMM weights)
  cvt(input,   in_bf,   ROWS_ * DM_);
  cvt(W_enc,   Wenc_bf, DM_ * DM_);
  cvt(W_in,    Win_bf,  2 * ED_ * DM_);
  cvt(W_x,     Wx_bf,   64 * ED_);
  cvt(W_dt,    Wdt_bf,  ED_ * DTR_);
  cvt(W_out,   Wout_bf, DM_ * ED_);
  cvt(W_outfc, Wofc_bf, DIN_ * DM_);
  cvt(W_mu,    Wmu_bf,  DOUT_ * DIN_);

  // 2) encode GEMM + bias + RMSNorm (fused), capture e_last
  encode_rmsnorm_kernel<<<dim3(ROWS_ / 16), T, 0, stream>>>(
      in_bf, Wenc_bf, b_enc, rms_w, u_bf, e_last);

  // 3) x_ = u @ W_in[:ED].T   (z half skipped for all but last step)
  gemm_bf16_wmma<<<dim3(ROWS_ / 16, ED_ / 128), T, 0, stream>>>(
      u_bf, DM_, Win_bf, nullptr, nullptr, x_pre, nullptr, ED_, DM_, 0);

  // 4) depthwise causal conv + SiLU
  conv_silu_kernel<<<dim3((ROWS_ * ED_) / 256), T, 0, stream>>>(
      x_pre, conv_w, conv_b, xs, xs_bf);

  // 5) dbc = xs @ W_x.T  (f32 for scan, bf16 copy feeds the dt GEMM)
  gemm_bf16_wmma<<<dim3(ROWS_ / 16, 1), T, 0, stream>>>(
      xs_bf, ED_, Wx_bf, nullptr, nullptr, dbc, dbc_bf, 64, ED_, 3);

  // 6) delta = softplus(dt @ W_dt.T + b_dt)   (dt = first 32 cols, lda=64)
  gemm_bf16_wmma<<<dim3(ROWS_ / 16, ED_ / 128), T, 0, stream>>>(
      dbc_bf, 64, Wdt_bf, b_dt, nullptr, delta, nullptr, ED_, DTR_, 1);

  // 7) fused selective scan (dA/dBu never materialized), y at t=L-1 only
  scan_kernel<<<dim3(B_ * 4), T, 0, stream>>>(delta, xs, dbc, A_log, D_, ylast);

  // 8) z gate at last step; y_bf = bf16(y * silu(z))
  zgate_kernel<<<dim3((B_ * ED_) / 256), T, 0, stream>>>(u_bf, Win_bf, ylast, y_bf);

  // 9) e_out = e_last + y @ W_out.T
  gemm_bf16_wmma<<<dim3(B_ / 16, (DM_ + 127) / 128), T, 0, stream>>>(
      y_bf, ED_, Wout_bf, nullptr, e_last, eout, eout_bf, DM_, ED_, 4);

  // 10) x = silu(e_out @ W_outfc.T + b_outfc)  -> d_out[0 : 32*512]
  gemm_bf16_wmma<<<dim3(B_ / 16, (DIN_ + 127) / 128), T, 0, stream>>>(
      eout_bf, DM_, Wofc_bf, b_outfc, nullptr, out, x_bf, DIN_, DM_, 2);

  // 11) mu = x @ W_mu.T + b_mu  -> d_out[32*512 : 32*512 + 32*64]
  gemm_bf16_wmma<<<dim3(B_ / 16, 1), T, 0, stream>>>(
      x_bf, DIN_, Wmu_bf, b_mu, nullptr, out + B_ * DIN_, nullptr, DOUT_, DIN_, 0);
}
